// TransformerPolicyNet_11458972746320
// MI455X (gfx1250) — matrix operand order repaired and checked by reading, not compiled, not run on hardware
//
#include <hip/hip_runtime.h>

// ---------------------------------------------------------------------------
// CDNA5 / gfx1250 bf16 WMMA transformer forward (compile-only tuned).
// Model constants (fixed by the reference):
#define D_MODEL 512
#define NHEADS  8
#define DHEAD   64
#define NLAYER  2
#define FFDIM   2048
#define NCTX    2048
#define NACT    2048
#define NSEQ    (NCTX + NACT)   // 4096
#define NGROUP  512
#define MAXPOS  32

typedef __attribute__((ext_vector_type(16))) __bf16 v16bf;
typedef __attribute__((ext_vector_type(8)))  __bf16 v8bf;
typedef __attribute__((ext_vector_type(8)))  float  v8f;
typedef __attribute__((ext_vector_type(4)))  float  v4f;

#define WMMA_BF16(a, b, c) \
  __builtin_amdgcn_wmma_f32_16x16x32_bf16(false, (a), false, (b), (short)0, (c), false, false)

// Two 16-byte LDS loads concatenated into one 16-element bf16 fragment.
__device__ __forceinline__ v16bf ld16(const __bf16* p0, const __bf16* p1) {
  v8bf lo = *(const v8bf*)p0;
  v8bf hi = *(const v8bf*)p1;
  return __builtin_shufflevector(lo, hi, 0, 1, 2, 3, 4, 5, 6, 7,
                                 8, 9, 10, 11, 12, 13, 14, 15);
}

// A fragment (16x32 bf16, row-major, leading dim ld; ld % 8 == 0):
// lane m = lane&15 holds row m; K-halves kh = 0/8; elems 0..7 at K=kh..kh+7,
// elems 8..15 at K=16+kh..16+kh+7 (ISA 7.12.2, 16-bit A 16x32).
__device__ __forceinline__ v16bf frag_a_load(const __bf16* __restrict__ t, int ld) {
  const int lane = threadIdx.x & 31;
  const __bf16* r0 = t + (lane & 15) * ld + ((lane >> 4) << 3);
  return ld16(r0, r0 + 16);
}

// B fragment (32x16) from K-major storage: column n lives at t + n*ld,
// K contiguous. Lane n = lane&15; kh = 0/8; elems 0..7 at K=kh..kh+7, elems 8..15 at K=16+kh..16+kh+7.
__device__ __forceinline__ v16bf frag_b_kmajor(const __bf16* __restrict__ t, int ld) {
  const int lane = threadIdx.x & 31;
  const __bf16* r0 = t + (lane & 15) * ld + ((lane >> 4) << 3);
  return ld16(r0, r0 + 16);
}

// C/D (16x16 f32): element r of v8f is (m = r + 8*(lane>>4), n = lane&15).

// ---------------------------------------------------------------------------
// GEMM: Out[M,N] = act(A[M,K] @ W[K,N] + bias[N]). M%128==0, N%128==0, K%32==0.
// 128x128 block tile, 8 waves (4 m x 2 n), each wave owns 32x64 (8 WMMAs/step).
#define BM 128
#define BN 128
#define BK 32
#define LDA 40   // padded, multiple of 8
#define LDK 40   // BsT leading dim (K direction), multiple of 8

__device__ __forceinline__ void store_tile(float* __restrict__ Out,
                                           const float* __restrict__ bias, v8f c,
                                           int N, int gm0, int gn, int relu) {
#pragma unroll
  for (int r = 0; r < 8; ++r) {
    float v = c[r] + bias[gn];
    if (relu) v = fmaxf(v, 0.f);
    Out[(size_t)(gm0 + r) * N + gn] = v;
  }
}

__global__ __launch_bounds__(256)
void gemm_bias_kernel(const float* __restrict__ A, const float* __restrict__ W,
                      const float* __restrict__ bias, float* __restrict__ Out,
                      int M, int N, int K, int relu) {
  __shared__ __bf16 As[BM * LDA];       // row-major rows x K
  __shared__ __bf16 BsT[BN * LDK];      // transposed: col n -> K-contiguous
  const int tid  = threadIdx.x;
  const int lane = tid & 31;
  const int wave = tid >> 5;
  const int wm = (wave & 3) * 32;       // 4 wave-rows  -> 128 rows
  const int wn = (wave >> 2) * 64;      // 2 wave-cols  -> 128 cols
  const int bm = blockIdx.y * BM;
  const int bn = blockIdx.x * BN;

  v8f acc00 = {}, acc01 = {}, acc02 = {}, acc03 = {};
  v8f acc10 = {}, acc11 = {}, acc12 = {}, acc13 = {};

  for (int k0 = 0; k0 < K; k0 += BK) {
    __syncthreads();
    // A tile: 128x32 f32 -> bf16 (1024 float4 / 256 thr).
#pragma unroll
    for (int i = 0; i < 4; ++i) {
      int idx4 = tid + i * 256;
      int r = idx4 >> 3, c4 = idx4 & 7;
      v4f v = *(const v4f*)&A[(size_t)(bm + r) * K + k0 + c4 * 4];
#pragma unroll
      for (int j = 0; j < 4; ++j) As[r * LDA + c4 * 4 + j] = (__bf16)v[j];
    }
    // W tile: 32x128 f32 -> bf16 transposed (1024 float4 / 256 thr).
#pragma unroll
    for (int i = 0; i < 4; ++i) {
      int idx4 = tid + i * 256;
      int r = idx4 >> 5, c4 = idx4 & 31;
      v4f v = *(const v4f*)&W[(size_t)(k0 + r) * N + bn + c4 * 4];
#pragma unroll
      for (int j = 0; j < 4; ++j) BsT[(c4 * 4 + j) * LDK + r] = (__bf16)v[j];
    }
    __syncthreads();
    v16bf a0 = frag_a_load(&As[wm * LDA], LDA);
    v16bf a1 = frag_a_load(&As[(wm + 16) * LDA], LDA);
    v16bf b0 = frag_b_kmajor(&BsT[(wn +  0) * LDK], LDK);
    v16bf b1 = frag_b_kmajor(&BsT[(wn + 16) * LDK], LDK);
    v16bf b2 = frag_b_kmajor(&BsT[(wn + 32) * LDK], LDK);
    v16bf b3 = frag_b_kmajor(&BsT[(wn + 48) * LDK], LDK);
    acc00 = WMMA_BF16(a0, b0, acc00);
    acc01 = WMMA_BF16(a0, b1, acc01);
    acc02 = WMMA_BF16(a0, b2, acc02);
    acc03 = WMMA_BF16(a0, b3, acc03);
    acc10 = WMMA_BF16(a1, b0, acc10);
    acc11 = WMMA_BF16(a1, b1, acc11);
    acc12 = WMMA_BF16(a1, b2, acc12);
    acc13 = WMMA_BF16(a1, b3, acc13);
  }

  const int rl = (lane >> 4) << 3;
  const int cl = lane & 15;
  const int gm0 = bm + wm + rl;
  const int gn0 = bn + wn + cl;
  store_tile(Out, bias, acc00, N, gm0,      gn0,      relu);
  store_tile(Out, bias, acc01, N, gm0,      gn0 + 16, relu);
  store_tile(Out, bias, acc02, N, gm0,      gn0 + 32, relu);
  store_tile(Out, bias, acc03, N, gm0,      gn0 + 48, relu);
  store_tile(Out, bias, acc10, N, gm0 + 16, gn0,      relu);
  store_tile(Out, bias, acc11, N, gm0 + 16, gn0 + 16, relu);
  store_tile(Out, bias, acc12, N, gm0 + 16, gn0 + 32, relu);
  store_tile(Out, bias, acc13, N, gm0 + 16, gn0 + 48, relu);
}

// ---------------------------------------------------------------------------
// Flash attention: one wave = (head h, 16 queries); keys streamed in chunks of
// 32 with online softmax. K kept row-major (Q*K^T reads K rows contiguously);
// V stored dh-major so the PV B-fragment is contiguous too.
#define LQK 72   // Qs/Ks leading dim (dh direction), multiple of 8
#define LVT 40   // VsT leading dim (key direction), multiple of 8

__global__ __launch_bounds__(32)
void flash_attn_kernel(const float* __restrict__ Q, const float* __restrict__ Kg,
                       const float* __restrict__ Vg, float* __restrict__ O,
                       int Lq, int Lk) {
  __shared__ __bf16 Qs[16 * LQK];     // [query][dh]
  __shared__ __bf16 Ks[32 * LQK];     // [key][dh]
  __shared__ __bf16 VsT[DHEAD * LVT]; // [dh][key]
  __shared__ __bf16 Ps[16 * 40];      // [query][key-chunk]
  __shared__ float  Sb[16 * 33];
  __shared__ float  Al[16];
  __shared__ float  El[16];

  const int lane = threadIdx.x;
  const int q0 = blockIdx.x * 16;
  const int h  = blockIdx.y;
  const float scale = 0.125f;   // dh^-0.5, dh = 64

  // Q tile: 16x64 (256 float4 / 32 lanes).
#pragma unroll
  for (int i = 0; i < 8; ++i) {
    int idx4 = lane + i * 32;
    int r = idx4 >> 4, c4 = idx4 & 15;
    v4f v = *(const v4f*)&Q[(size_t)(q0 + r) * D_MODEL + h * DHEAD + c4 * 4];
#pragma unroll
    for (int j = 0; j < 4; ++j) Qs[r * LQK + c4 * 4 + j] = (__bf16)(v[j] * scale);
  }
  __syncthreads();
  v16bf aq0 = frag_a_load(Qs, LQK);        // dh 0..31
  v16bf aq1 = frag_a_load(Qs + 32, LQK);   // dh 32..63

  float mrow = -3.0e38f, lrow = 0.f;
  v8f o0 = {}, o1 = {}, o2 = {}, o3 = {};
  const int rl = (lane >> 4) << 3;
  const int cl = lane & 15;

  for (int kb0 = 0; kb0 < Lk; kb0 += 32) {
    __syncthreads();
    // K/V chunk: 32x64 each (512 float4 / 32 lanes).
#pragma unroll
    for (int i = 0; i < 16; ++i) {
      int idx4 = lane + i * 32;
      int r = idx4 >> 4, c4 = idx4 & 15;
      size_t goff = (size_t)(kb0 + r) * D_MODEL + h * DHEAD + c4 * 4;
      v4f kv = *(const v4f*)&Kg[goff];
      v4f vv = *(const v4f*)&Vg[goff];
#pragma unroll
      for (int j = 0; j < 4; ++j) {
        Ks[r * LQK + c4 * 4 + j] = (__bf16)kv[j];
        VsT[(c4 * 4 + j) * LVT + r] = (__bf16)vv[j];
      }
    }
    __syncthreads();
    // S(16x32) = Q(16x64) * K^T ; B[d][n] = Ks[n_row][d] -> K-major columns.
#pragma unroll
    for (int nt = 0; nt < 2; ++nt) {
      v8f s = {};
      v16bf bk0 = frag_b_kmajor(&Ks[(nt * 16) * LQK], LQK);        // d = 0..31
      v16bf bk1 = frag_b_kmajor(&Ks[(nt * 16) * LQK] + 32, LQK);   // d = 32..63
      s = WMMA_BF16(aq0, bk0, s);
      s = WMMA_BF16(aq1, bk1, s);
#pragma unroll
      for (int r = 0; r < 8; ++r) Sb[(rl + r) * 33 + nt * 16 + cl] = s[r];
    }
    __syncthreads();
    if (lane < 16) {
      float cm = -3.0e38f;
#pragma unroll
      for (int j = 0; j < 32; ++j) cm = fmaxf(cm, Sb[lane * 33 + j]);
      float nm = fmaxf(mrow, cm);
      float al = __expf(mrow - nm);
      float ps = 0.f;
#pragma unroll
      for (int j = 0; j < 32; ++j) {
        float p = __expf(Sb[lane * 33 + j] - nm);
        ps += p;
        Ps[lane * 40 + j] = (__bf16)p;
      }
      lrow = lrow * al + ps;
      mrow = nm;
      Al[lane] = al;
    }
    __syncthreads();
#pragma unroll
    for (int r = 0; r < 8; ++r) {
      float al = Al[rl + r];
      o0[r] *= al; o1[r] *= al; o2[r] *= al; o3[r] *= al;
    }
    v16bf ap  = frag_a_load(Ps, 40);                       // P 16x32
    v16bf bv0 = frag_b_kmajor(&VsT[0 * 16 * LVT], LVT);    // dh  0..15
    v16bf bv1 = frag_b_kmajor(&VsT[1 * 16 * LVT], LVT);    // dh 16..31
    v16bf bv2 = frag_b_kmajor(&VsT[2 * 16 * LVT], LVT);    // dh 32..47
    v16bf bv3 = frag_b_kmajor(&VsT[3 * 16 * LVT], LVT);    // dh 48..63
    o0 = WMMA_BF16(ap, bv0, o0);
    o1 = WMMA_BF16(ap, bv1, o1);
    o2 = WMMA_BF16(ap, bv2, o2);
    o3 = WMMA_BF16(ap, bv3, o3);
  }
  if (lane < 16) El[lane] = lrow;
  __syncthreads();
#pragma unroll
  for (int r = 0; r < 8; ++r) {
    float inv = 1.f / El[rl + r];
    float* orow = O + (size_t)(q0 + rl + r) * D_MODEL + h * DHEAD;
    orow[0 * 16 + cl] = o0[r] * inv;
    orow[1 * 16 + cl] = o1[r] * inv;
    orow[2 * 16 + cl] = o2[r] * inv;
    orow[3 * 16 + cl] = o3[r] * inv;
  }
}

// ---------------------------------------------------------------------------
__global__ __launch_bounds__(D_MODEL)
void add_embed_kernel(float* __restrict__ emb, const float* __restrict__ temb,
                      const float* __restrict__ pemb, const int* __restrict__ type_ids,
                      const int* __restrict__ pos_ids) {
  const int row = blockIdx.x;
  const int c = threadIdx.x;
  int t = min(max(type_ids[row], 0), 1);
  float v = emb[(size_t)row * D_MODEL + c] + temb[t * D_MODEL + c];
  if (row >= NCTX) {
    int p = min(max(pos_ids[row], 0), MAXPOS - 1);
    v += pemb[p * D_MODEL + c];
  }
  emb[(size_t)row * D_MODEL + c] = v;
}

// X = LayerNorm(X + R) * g + b  (row-wise over D=512)
__global__ __launch_bounds__(128)
void add_ln_kernel(float* __restrict__ X, const float* __restrict__ R,
                   const float* __restrict__ g, const float* __restrict__ b) {
  __shared__ float red[128], red2[128];
  const int row = blockIdx.x;
  const int tid = threadIdx.x;
  float vals[4];
  float s = 0.f, s2 = 0.f;
#pragma unroll
  for (int i = 0; i < 4; ++i) {
    int c = tid + i * 128;
    float v = X[(size_t)row * D_MODEL + c] + R[(size_t)row * D_MODEL + c];
    vals[i] = v; s += v; s2 += v * v;
  }
  red[tid] = s; red2[tid] = s2;
  __syncthreads();
  for (int off = 64; off > 0; off >>= 1) {
    if (tid < off) { red[tid] += red[tid + off]; red2[tid] += red2[tid + off]; }
    __syncthreads();
  }
  float mu  = red[0] * (1.f / D_MODEL);
  float var = red2[0] * (1.f / D_MODEL) - mu * mu;
  float inv = rsqrtf(var + 1e-5f);
#pragma unroll
  for (int i = 0; i < 4; ++i) {
    int c = tid + i * 128;
    X[(size_t)row * D_MODEL + c] = (vals[i] - mu) * inv * g[c] + b[c];
  }
}

__global__ void zero_kernel(float* __restrict__ p, int n) {
  int i = blockIdx.x * blockDim.x + threadIdx.x;
  if (i < n) p[i] = 0.f;
}

__global__ __launch_bounds__(D_MODEL)
void scatter_kernel(const float* __restrict__ act, const int* __restrict__ gid,
                    float* __restrict__ pooled, float* __restrict__ counts) {
  const int a = blockIdx.x;
  const int g = gid[a];
  atomicAdd(&pooled[(size_t)g * D_MODEL + threadIdx.x],
            act[(size_t)a * D_MODEL + threadIdx.x]);
  if (threadIdx.x == 0) atomicAdd(&counts[g], 1.0f);
}

__global__ __launch_bounds__(D_MODEL)
void divide_kernel(float* __restrict__ pooled, const float* __restrict__ counts) {
  const int gidx = blockIdx.x;
  float c = fmaxf(counts[gidx], 1.0f);
  pooled[(size_t)gidx * D_MODEL + threadIdx.x] /= c;
}

__global__ void head2_kernel(const float* __restrict__ hb, const float* __restrict__ W2,
                             const float* __restrict__ b2, float* __restrict__ out) {
  int gidx = blockIdx.x * blockDim.x + threadIdx.x;
  if (gidx < NGROUP) {
    float s = b2[0];
    for (int i = 0; i < D_MODEL; ++i) s += hb[(size_t)gidx * D_MODEL + i] * W2[i];
    out[gidx] = s;
  }
}

// ---------------------------------------------------------------------------
// Host orchestration. Input order = setup_inputs() dict flattening:
//  0 feats_ctx 1 feats_act 2 pWc 3 pbc 4 pWa 5 pba 6 temb 7 pemb
//  enc:   8 Wq 9 bq 10 Wk 11 bk 12 Wv 13 bv 14 Wo 15 bo
//         16 ln1g 17 ln1b 18 ln2g 19 ln2b 20 fW1 21 fb1 22 fW2 23 fb2
//  cross: 24 sWq..31 sbo, 32 cWq..39 cbo, 40 ln1g..45 ln3b, 46 fW1..49 fb2
//  50 head.W1 51 head.b1 52 head.W2 53 head.b2
//  54 type_ids 55 pos_ids 56 group_ids 57 num_context 58 num_groups
extern "C" void kernel_launch(void* const* d_in, const int* in_sizes, int n_in,
                              void* d_out, int out_size, void* d_ws, size_t ws_size,
                              hipStream_t stream) {
  (void)in_sizes; (void)n_in; (void)out_size; (void)ws_size;
  const float* feats_ctx = (const float*)d_in[0];
  const float* feats_act = (const float*)d_in[1];
  const float* pWc = (const float*)d_in[2];
  const float* pbc = (const float*)d_in[3];
  const float* pWa = (const float*)d_in[4];
  const float* pba = (const float*)d_in[5];
  const float* temb = (const float*)d_in[6];
  const float* pemb = (const float*)d_in[7];
  const float* eW[4]  = {(const float*)d_in[8],  (const float*)d_in[10],
                         (const float*)d_in[12], (const float*)d_in[14]};
  const float* eB[4]  = {(const float*)d_in[9],  (const float*)d_in[11],
                         (const float*)d_in[13], (const float*)d_in[15]};
  const float* eln1g = (const float*)d_in[16]; const float* eln1b = (const float*)d_in[17];
  const float* eln2g = (const float*)d_in[18]; const float* eln2b = (const float*)d_in[19];
  const float* efW1 = (const float*)d_in[20]; const float* efb1 = (const float*)d_in[21];
  const float* efW2 = (const float*)d_in[22]; const float* efb2 = (const float*)d_in[23];
  const float* sW[4]  = {(const float*)d_in[24], (const float*)d_in[26],
                         (const float*)d_in[28], (const float*)d_in[30]};
  const float* sB[4]  = {(const float*)d_in[25], (const float*)d_in[27],
                         (const float*)d_in[29], (const float*)d_in[31]};
  const float* cW[4]  = {(const float*)d_in[32], (const float*)d_in[34],
                         (const float*)d_in[36], (const float*)d_in[38]};
  const float* cB[4]  = {(const float*)d_in[33], (const float*)d_in[35],
                         (const float*)d_in[37], (const float*)d_in[39]};
  const float* xln1g = (const float*)d_in[40]; const float* xln1b = (const float*)d_in[41];
  const float* xln2g = (const float*)d_in[42]; const float* xln2b = (const float*)d_in[43];
  const float* xln3g = (const float*)d_in[44]; const float* xln3b = (const float*)d_in[45];
  const float* xfW1 = (const float*)d_in[46]; const float* xfb1 = (const float*)d_in[47];
  const float* xfW2 = (const float*)d_in[48]; const float* xfb2 = (const float*)d_in[49];
  const float* hW1 = (const float*)d_in[50]; const float* hb1 = (const float*)d_in[51];
  const float* hW2 = (const float*)d_in[52]; const float* hb2 = (const float*)d_in[53];
  const int* type_ids  = (const int*)d_in[54];
  const int* pos_ids   = (const int*)d_in[55];
  const int* group_ids = (const int*)d_in[56];

  float* ws = (float*)d_ws;
  float* emb    = ws;                                  // NSEQ x D
  float* qbuf   = emb    + (size_t)NSEQ * D_MODEL;     // NACT x D
  float* kbuf   = qbuf   + (size_t)NACT * D_MODEL;     // NSEQ x D
  float* vbuf   = kbuf   + (size_t)NSEQ * D_MODEL;     // NSEQ x D
  float* attnb  = vbuf   + (size_t)NSEQ * D_MODEL;     // NACT x D
  float* tmpb   = attnb  + (size_t)NACT * D_MODEL;     // NACT x D
  float* ffh    = tmpb   + (size_t)NACT * D_MODEL;     // NACT x FF
  float* pooled = ffh    + (size_t)NACT * FFDIM;       // G x D
  float* cnts   = pooled + (size_t)NGROUP * D_MODEL;   // G
  float* hbuf   = cnts   + NGROUP;                     // G x D

  float* ctx = emb;
  float* act = emb + (size_t)NCTX * D_MODEL;

  auto gemm = [&](const float* A, const float* W, const float* bias, float* Out,
                  int M, int N, int K, int relu) {
    dim3 grid(N / BN, M / BM);
    gemm_bias_kernel<<<grid, 256, 0, stream>>>(A, W, bias, Out, M, N, K, relu);
  };
  auto flash = [&](const float* Qp, const float* Kp, const float* Vp, float* Op,
                   int Lq, int Lk) {
    dim3 grid(Lq / 16, NHEADS);
    flash_attn_kernel<<<grid, 32, 0, stream>>>(Qp, Kp, Vp, Op, Lq, Lk);
  };
  auto addln = [&](float* X, const float* R, const float* g, const float* b, int M) {
    add_ln_kernel<<<M, 128, 0, stream>>>(X, R, g, b);
  };

  // ---- embeddings ----
  gemm(feats_ctx, pWc, pbc, ctx, NCTX, D_MODEL, 128, 0);
  gemm(feats_act, pWa, pba, act, NACT, D_MODEL, 64, 0);
  add_embed_kernel<<<NSEQ, D_MODEL, 0, stream>>>(emb, temb, pemb, type_ids, pos_ids);

  const size_t DD = (size_t)D_MODEL * D_MODEL;
  // ---- encoder (ctx self-attention) ----
  for (int i = 0; i < NLAYER; ++i) {
    gemm(ctx, eW[0] + i * DD, eB[0] + i * D_MODEL, qbuf, NCTX, D_MODEL, D_MODEL, 0);
    gemm(ctx, eW[1] + i * DD, eB[1] + i * D_MODEL, kbuf, NCTX, D_MODEL, D_MODEL, 0);
    gemm(ctx, eW[2] + i * DD, eB[2] + i * D_MODEL, vbuf, NCTX, D_MODEL, D_MODEL, 0);
    flash(qbuf, kbuf, vbuf, attnb, NCTX, NCTX);
    gemm(attnb, eW[3] + i * DD, eB[3] + i * D_MODEL, tmpb, NCTX, D_MODEL, D_MODEL, 0);
    addln(ctx, tmpb, eln1g + i * D_MODEL, eln1b + i * D_MODEL, NCTX);
    gemm(ctx, efW1 + (size_t)i * D_MODEL * FFDIM, efb1 + i * FFDIM, ffh, NCTX, FFDIM, D_MODEL, 1);
    gemm(ffh, efW2 + (size_t)i * FFDIM * D_MODEL, efb2 + i * D_MODEL, tmpb, NCTX, D_MODEL, FFDIM, 0);
    addln(ctx, tmpb, eln2g + i * D_MODEL, eln2b + i * D_MODEL, NCTX);
  }

  // ---- decoder (act self + cross) ----
  for (int i = 0; i < NLAYER; ++i) {
    // self-attention
    gemm(act, sW[0] + i * DD, sB[0] + i * D_MODEL, qbuf, NACT, D_MODEL, D_MODEL, 0);
    gemm(act, sW[1] + i * DD, sB[1] + i * D_MODEL, kbuf, NACT, D_MODEL, D_MODEL, 0);
    gemm(act, sW[2] + i * DD, sB[2] + i * D_MODEL, vbuf, NACT, D_MODEL, D_MODEL, 0);
    flash(qbuf, kbuf, vbuf, attnb, NACT, NACT);
    gemm(attnb, sW[3] + i * DD, sB[3] + i * D_MODEL, tmpb, NACT, D_MODEL, D_MODEL, 0);
    addln(act, tmpb, xln1g + i * D_MODEL, xln1b + i * D_MODEL, NACT);
    // cross-attention: kv = concat(ctx, act) -> project both halves
    gemm(act, cW[0] + i * DD, cB[0] + i * D_MODEL, qbuf, NACT, D_MODEL, D_MODEL, 0);
    gemm(ctx, cW[1] + i * DD, cB[1] + i * D_MODEL, kbuf, NCTX, D_MODEL, D_MODEL, 0);
    gemm(act, cW[1] + i * DD, cB[1] + i * D_MODEL, kbuf + (size_t)NCTX * D_MODEL, NACT, D_MODEL, D_MODEL, 0);
    gemm(ctx, cW[2] + i * DD, cB[2] + i * D_MODEL, vbuf, NCTX, D_MODEL, D_MODEL, 0);
    gemm(act, cW[2] + i * DD, cB[2] + i * D_MODEL, vbuf + (size_t)NCTX * D_MODEL, NACT, D_MODEL, D_MODEL, 0);
    flash(qbuf, kbuf, vbuf, attnb, NACT, NSEQ);
    gemm(attnb, cW[3] + i * DD, cB[3] + i * D_MODEL, tmpb, NACT, D_MODEL, D_MODEL, 0);
    addln(act, tmpb, xln2g + i * D_MODEL, xln2b + i * D_MODEL, NACT);
    // FFN
    gemm(act, xfW1 + (size_t)i * D_MODEL * FFDIM, xfb1 + i * FFDIM, ffh, NACT, FFDIM, D_MODEL, 1);
    gemm(ffh, xfW2 + (size_t)i * FFDIM * D_MODEL, xfb2 + i * D_MODEL, tmpb, NACT, D_MODEL, FFDIM, 0);
    addln(act, tmpb, xln3g + i * D_MODEL, xln3b + i * D_MODEL, NACT);
  }

  // ---- pooling + head ----
  int zn = NGROUP * D_MODEL + NGROUP;
  zero_kernel<<<(zn + 255) / 256, 256, 0, stream>>>(pooled, zn);
  scatter_kernel<<<NACT, D_MODEL, 0, stream>>>(act, group_ids, pooled, cnts);
  divide_kernel<<<NGROUP, D_MODEL, 0, stream>>>(pooled, cnts);
  gemm(pooled, hW1, hb1, hbuf, NGROUP, D_MODEL, D_MODEL, 1);
  head2_kernel<<<2, 256, 0, stream>>>(hbuf, hW2, hb2, (float*)d_out);
}
